// ParameterMemoryBank_75831942578466
// MI455X (gfx1250) — compile-verified, matrix operand order checked by reading, not run on hardware
//
#include <hip/hip_runtime.h>
#include <hip/hip_bf16.h>

#define NUM_BLOCKS 32
#define BLOCK_CAP  1024
#define KEY_DIM    256
#define VALUE_DIM  256
#define HIDDEN     768
#define LN_EPS     1e-5f

typedef __attribute__((ext_vector_type(16))) __bf16 v16bf;
typedef __attribute__((ext_vector_type(8)))  __bf16 v8bf;
typedef __attribute__((ext_vector_type(8)))  float  v8f;

// Load one 16-element bf16 WMMA fragment (A or B operand) for this lane.
// p is already offset by this lane's row and klo; elements are p[0..7] and
// p[16..23] per the 16-bit A/B VGPR layout (ISA 7.12.2).
__device__ __forceinline__ v16bf load_frag_bf16(const __bf16* p) {
  v8bf lo = *(const v8bf*)(p);
  v8bf hi = *(const v8bf*)(p + 16);
  v16bf r;
#pragma unroll
  for (int i = 0; i < 8; ++i) { r[i] = lo[i]; r[i + 8] = hi[i]; }
  return r;
}

// CDNA5 async global->LDS copy, 16B per lane, tracked by ASYNCcnt.
__device__ __forceinline__ void async_b128(unsigned lds_off, const void* g) {
  asm volatile("global_load_async_to_lds_b128 %0, %1, off"
               :: "v"(lds_off), "v"(g) : "memory");
}

// ---------------- stage 0: zero the retrieved accumulator ----------------
__global__ void zero_f32_kernel(float* p, int n) {
  for (int i = blockIdx.x * blockDim.x + threadIdx.x; i < n;
       i += gridDim.x * blockDim.x)
    p[i] = 0.0f;
}

// ---------------- stage 1: keys f32 -> bf16 (same layout) ----------------
__global__ void cvt_bf16_kernel(__bf16* o, const float* in, int n) {
  for (int i = blockIdx.x * blockDim.x + threadIdx.x; i < n;
       i += gridDim.x * blockDim.x)
    o[i] = (__bf16)in[i];
}

// ------- stage 2: values[b][c][d] f32 -> values_t[b][d][c] bf16 ----------
__global__ void transpose_vals_kernel(__bf16* vt, const float* v, int n) {
  for (int i = blockIdx.x * blockDim.x + threadIdx.x; i < n;
       i += gridDim.x * blockDim.x) {
    int c = i & (BLOCK_CAP - 1);
    int d = (i >> 10) & (VALUE_DIM - 1);
    int b = i >> 18;
    vt[i] = (__bf16)v[((size_t)b * BLOCK_CAP + c) * VALUE_DIM + d];
  }
}

// -------- stage 3: q = LayerNorm(hidden @ Wk + bk) -> bf16 [N,256] -------
__global__ void qproj_ln_kernel(__bf16* qbf, const float* hs, const float* Wk,
                                const float* bk, const float* lnw,
                                const float* lnb) {
  __shared__ float sh[8][HIDDEN];
  __shared__ float redA[256], redB[256];
  const int tid = threadIdx.x;
  const int row0 = blockIdx.x * 8;
  for (int idx = tid; idx < 8 * HIDDEN; idx += 256) {
    int r = idx / HIDDEN, h = idx - r * HIDDEN;
    sh[r][h] = hs[(size_t)(row0 + r) * HIDDEN + h];
  }
  __syncthreads();
  float acc[8];
#pragma unroll
  for (int r = 0; r < 8; ++r) acc[r] = 0.0f;
  const int d = tid;  // KEY_DIM == blockDim.x == 256
  for (int h = 0; h < HIDDEN; ++h) {
    float w = Wk[(size_t)h * KEY_DIM + d];
#pragma unroll
    for (int r = 0; r < 8; ++r) acc[r] += sh[r][h] * w;
  }
  const float bkd = bk[d], lw = lnw[d], lb = lnb[d];
  for (int r = 0; r < 8; ++r) {
    float v = acc[r] + bkd;
    redA[tid] = v;
    redB[tid] = v * v;
    __syncthreads();
    for (int s = 128; s > 0; s >>= 1) {
      if (tid < s) { redA[tid] += redA[tid + s]; redB[tid] += redB[tid + s]; }
      __syncthreads();
    }
    float mu = redA[0] * (1.0f / KEY_DIM);
    float var = redB[0] * (1.0f / KEY_DIM) - mu * mu;
    float q = (v - mu) * rsqrtf(var + LN_EPS) * lw + lb;
    qbf[(size_t)(row0 + r) * KEY_DIM + d] = (__bf16)q;
    __syncthreads();
  }
}

// ---------------- stage 4: per-block flash attention (WMMA) --------------
// Transposed scheme: S^T = K @ Q^T, O^T = V^T @ P^T. grid (N/128, 32),
// 256 threads = 8 waves, wave owns 16 query columns. K/V chunks are staged
// into double-buffered LDS with async global->LDS copies (ASYNCcnt).
__global__ void __launch_bounds__(256)
attn_kernel(float* retr_t, const __bf16* qbf, const __bf16* kbf,
            const __bf16* vt, const int* usage, int N) {
  // padded rows: 264/40 elems keep b128 lane reads on disjoint bank groups
  __shared__ __align__(16) __bf16 kbuf[2][32][264];
  __shared__ __align__(16) __bf16 vbuf[2][256][40];

  const int tid = threadIdx.x;
  const int lane = tid & 31;
  const int w = tid >> 5;
  const int b = blockIdx.y;
  const int qrow0 = blockIdx.x * 128 + w * 16;
  const int use = usage[b];
  const int nch = (use + 31) >> 5;  // chunks of 32 keys; use >= 1
  const int hi = lane >> 4;
  const int ln = lane & 15;
  const int klo = hi << 3;

  const __bf16* qrow = qbf + (size_t)(qrow0 + ln) * KEY_DIM + klo;
  const __bf16* kblk = kbf + (size_t)b * BLOCK_CAP * KEY_DIM;
  const __bf16* vblk = vt + (size_t)b * VALUE_DIM * BLOCK_CAP;

  // cooperative async staging of one 32-key chunk (K: 16KB, V: 16KB)
  auto stage = [&](int bi, int c0) {
    const __bf16* kchunk = kblk + (size_t)c0 * KEY_DIM;
    const __bf16* vchunk = vblk + c0;
#pragma unroll
    for (int j = 0; j < 4; ++j) {
      int u = tid + j * 256;  // 16B units; K row = u>>5, col = (u&31)*8
      unsigned lds = (unsigned)(unsigned long long)
          &kbuf[bi][u >> 5][(u & 31) * 8];
      async_b128(lds, kchunk + u * 8);
    }
#pragma unroll
    for (int j = 0; j < 4; ++j) {
      int u = tid + j * 256;  // V row (d) = u>>2, part = u&3
      unsigned lds = (unsigned)(unsigned long long)
          &vbuf[bi][u >> 2][(u & 3) * 8];
      async_b128(lds, vchunk + (size_t)(u >> 2) * BLOCK_CAP + (u & 3) * 8);
    }
  };

  // Q fragments (B operand of S^T) live in registers for the whole kernel.
  v16bf qa[8];
#pragma unroll
  for (int s = 0; s < 8; ++s) qa[s] = load_frag_bf16(qrow + 32 * s);

  v8f o[16];
#pragma unroll
  for (int dt = 0; dt < 16; ++dt) o[dt] = (v8f){0, 0, 0, 0, 0, 0, 0, 0};
  float mrun = -__builtin_inff(), lrun = 0.0f;

  stage(0, 0);
  asm volatile("s_wait_asynccnt 0x0" ::: "memory");
  __syncthreads();

  for (int ch = 0; ch < nch; ++ch) {
    const int cur = ch & 1;
    const int col0 = ch * 32;
    if (ch + 1 < nch) stage((ch + 1) & 1, col0 + 32);  // prefetch next

    // ---- S^T tiles: ct0 = keys col0..+15, ct1 = keys col0+16..+31 ----
    v8f ct0 = (v8f){0, 0, 0, 0, 0, 0, 0, 0};
    v8f ct1 = (v8f){0, 0, 0, 0, 0, 0, 0, 0};
#pragma unroll
    for (int s = 0; s < 8; ++s) {
      v16bf ka0 = load_frag_bf16(&kbuf[cur][ln][32 * s + klo]);
      v16bf ka1 = load_frag_bf16(&kbuf[cur][16 + ln][32 * s + klo]);
      ct0 = __builtin_amdgcn_wmma_f32_16x16x32_bf16(false, ka0, false, qa[s],
                                                    (short)0, ct0, false,
                                                    false);
      ct1 = __builtin_amdgcn_wmma_f32_16x16x32_bf16(false, ka1, false, qa[s],
                                                    (short)0, ct1, false,
                                                    false);
    }
    ct0 = ct0 * 0.0625f;  // 1/sqrt(KEY_DIM)
    ct1 = ct1 * 0.0625f;
    // mask: lane holds key rows col0 + hi*8 + r (tile0), +16 (tile1)
#pragma unroll
    for (int r = 0; r < 8; ++r) {
      ct0[r] = (col0 + hi * 8 + r < use) ? ct0[r] : -__builtin_inff();
      ct1[r] = (col0 + 16 + hi * 8 + r < use) ? ct1[r] : -__builtin_inff();
    }
    // ---- online softmax over keys for this lane's query column ----
    float mx = fmaxf(ct0[0], ct1[0]);
#pragma unroll
    for (int r = 1; r < 8; ++r) mx = fmaxf(mx, fmaxf(ct0[r], ct1[r]));
    mx = fmaxf(mx, __shfl_xor(mx, 16, 32));  // other key-half of column
    float nm = fmaxf(mrun, mx);
    float sc = __expf(mrun - nm);
    mrun = nm;
    float rs = 0.0f;
#pragma unroll
    for (int r = 0; r < 8; ++r) {
      float e0 = __expf(ct0[r] - nm);
      float e1 = __expf(ct1[r] - nm);
      ct0[r] = e0;
      ct1[r] = e1;
      rs += e0 + e1;
    }
    rs += __shfl_xor(rs, 16, 32);
    lrun = lrun * sc + rs;
    // ---- P^T -> B fragment: lane-local repack (no LDS transpose) ----
    v16bf pb;
#pragma unroll
    for (int r = 0; r < 8; ++r) {
      pb[r] = (__bf16)ct0[r];
      pb[r + 8] = (__bf16)ct1[r];
    }
#pragma unroll
    for (int dt = 0; dt < 16; ++dt) o[dt] = o[dt] * sc;
    // ---- O^T += V^T @ P^T over this chunk (16 d-tiles) ----
#pragma unroll
    for (int dt = 0; dt < 16; ++dt) {
      v16bf va = load_frag_bf16(&vbuf[cur][dt * 16 + ln][klo]);
      o[dt] = __builtin_amdgcn_wmma_f32_16x16x32_bf16(false, va, false, pb,
                                                      (short)0, o[dt], false,
                                                      false);
    }
    asm volatile("s_wait_asynccnt 0x0" ::: "memory");  // next chunk landed
    __syncthreads();
  }
  // ---- normalize and accumulate (d-major layout, lanes coalesced) ----
  float inv = 1.0f / lrun;
#pragma unroll
  for (int dt = 0; dt < 16; ++dt)
#pragma unroll
    for (int r = 0; r < 8; ++r)
      atomicAdd(&retr_t[(size_t)(dt * 16 + hi * 8 + r) * N + qrow0 + ln],
                o[dt][r] * inv);
}

// ---------- stage 5: out = retrieved^T @ Wo + bo (retr_t is d-major) -----
__global__ void outproj_kernel(float* out, const float* retr_t,
                               const float* Wo, const float* bo, int N) {
  __shared__ float sh[8][VALUE_DIM];
  const int tid = threadIdx.x;
  const int row0 = blockIdx.x * 8;
  for (int idx = tid; idx < 8 * VALUE_DIM; idx += 256) {
    int h = idx >> 3, r = idx & 7;
    sh[r][h] = retr_t[(size_t)h * N + row0 + r];
  }
  __syncthreads();
  float acc[3][8];
#pragma unroll
  for (int j = 0; j < 3; ++j)
#pragma unroll
    for (int r = 0; r < 8; ++r) acc[j][r] = 0.0f;
  for (int h = 0; h < VALUE_DIM; ++h) {
#pragma unroll
    for (int j = 0; j < 3; ++j) {
      float wv = Wo[(size_t)h * HIDDEN + j * 256 + tid];
#pragma unroll
      for (int r = 0; r < 8; ++r) acc[j][r] += sh[r][h] * wv;
    }
  }
#pragma unroll
  for (int j = 0; j < 3; ++j) {
    float bv = bo[j * 256 + tid];
#pragma unroll
    for (int r = 0; r < 8; ++r)
      out[(size_t)(row0 + r) * HIDDEN + j * 256 + tid] = acc[j][r] + bv;
  }
}

extern "C" void kernel_launch(void* const* d_in, const int* in_sizes, int n_in,
                              void* d_out, int out_size, void* d_ws,
                              size_t ws_size, hipStream_t stream) {
  const float* hs   = (const float*)d_in[0];
  const float* Wk   = (const float*)d_in[1];
  const float* bk   = (const float*)d_in[2];
  const float* lnw  = (const float*)d_in[3];
  const float* lnb  = (const float*)d_in[4];
  const float* mk   = (const float*)d_in[5];
  const float* mv   = (const float*)d_in[6];
  const float* Wo   = (const float*)d_in[7];
  const float* bo   = (const float*)d_in[8];
  const int* usage  = (const int*)d_in[9];
  float* out = (float*)d_out;

  const int N = in_sizes[0] / HIDDEN;  // B*S = 8192
  const int nkv = NUM_BLOCKS * BLOCK_CAP * KEY_DIM;

  char* ws = (char*)d_ws;
  size_t off = 0;
  __bf16* qbf = (__bf16*)(ws + off); off += (size_t)N * KEY_DIM * 2;
  __bf16* kbf = (__bf16*)(ws + off); off += (size_t)nkv * 2;
  __bf16* vtb = (__bf16*)(ws + off); off += (size_t)nkv * 2;
  float* retr = (float*)(ws + off);  // [VALUE_DIM][N] floats, d-major

  zero_f32_kernel<<<1024, 256, 0, stream>>>(retr, N * VALUE_DIM);
  cvt_bf16_kernel<<<2048, 256, 0, stream>>>(kbf, mk, nkv);
  transpose_vals_kernel<<<2048, 256, 0, stream>>>(vtb, mv, nkv);
  qproj_ln_kernel<<<N / 8, 256, 0, stream>>>(qbf, hs, Wk, bk, lnw, lnb);
  attn_kernel<<<dim3(N / 128, NUM_BLOCKS), 256, 0, stream>>>(retr, qbf, kbf,
                                                             vtb, usage, N);
  outproj_kernel<<<N / 8, 256, 0, stream>>>(out, retr, Wo, bo, N);
}
